// LinearAttention_79362405695749
// MI455X (gfx1250) — matrix operand order repaired and checked by reading, not compiled, not run on hardware
//
#include <hip/hip_runtime.h>
#include <hip/hip_bf16.h>
#include <stdint.h>

// MI455X (gfx1250) linear-attention pipeline: wave32, WMMA bf16, TDM async tiles.
typedef __attribute__((ext_vector_type(16))) __bf16 v16bf;
typedef __attribute__((ext_vector_type(8)))  __bf16 v8bf;
typedef __attribute__((ext_vector_type(4)))  __bf16 v4bf;
typedef __attribute__((ext_vector_type(8)))  float  v8f;
typedef __attribute__((ext_vector_type(4)))  unsigned int su4;
typedef __attribute__((ext_vector_type(8)))  int si8;
typedef __attribute__((ext_vector_type(4)))  int si4;

#define B_   4
#define N_   4096
#define C_   1024
#define H_   16
#define D_   64
#define NQK  (2*C_)      // 2048

static __device__ __forceinline__ v16bf loadA16(const __bf16* p) {
  // A-fragment per-lane data: two contiguous 8-element (16B) chunks.
  v8bf lo = *reinterpret_cast<const v8bf*>(p);
  v8bf hi = *reinterpret_cast<const v8bf*>(p + 16);
  return __builtin_shufflevector(lo, hi, 0,1,2,3,4,5,6,7,8,9,10,11,12,13,14,15);
}

static __device__ __forceinline__ v8f wmma_bf16(v16bf a, v16bf b, v8f c) {
  return __builtin_amdgcn_wmma_f32_16x16x32_bf16(false, a, false, b,
                                                 (short)0, c, false, false);
}

// ---------------------------------------------------------------------------
// Tensor Data Mover: async 2D tile (global -> LDS), bf16 elements.
//   g1w0 carries data_size / pad_enable / pad_interval / pad_amount.
//   Tile == tensor dims (no OOB), row stride `stride0` elements in global.
// ---------------------------------------------------------------------------
static __device__ __forceinline__ void tdm_load_2d(uint32_t lds_addr,
                                                   const void* gptr,
                                                   uint32_t g1w0,
                                                   uint32_t dim0, uint32_t dim1,
                                                   uint32_t stride0) {
  uint64_t ga = (uint64_t)(uintptr_t)gptr;
  su4 g0;
  g0.x = 1u;                                                   // count=1, user mode
  g0.y = lds_addr;                                             // LDS byte address
  g0.z = (uint32_t)ga;                                         // global addr [31:0]
  g0.w = ((uint32_t)(ga >> 32) & 0x1FFFFFFu) | (2u << 30);     // [56:32] | type=2
  si8 g1;
  g1[0] = (int)g1w0;                                           // mask|dsize|pad cfg
  g1[1] = (int)(dim0 << 16);                                   // tensor_dim0 lo16
  g1[2] = (int)((dim0 >> 16) | (dim1 << 16));                  // d0 hi | d1 lo
  g1[3] = (int)((dim1 >> 16) | (dim0 << 16));                  // d1 hi | tile_dim0
  g1[4] = (int)(dim1 & 0xFFFFu);                               // tile_dim1 (dim2=0)
  g1[5] = (int)stride0;                                        // dim0 stride lo32
  g1[6] = 0;
  g1[7] = 0;
  si4 z4 = {0, 0, 0, 0};
#if defined(__clang_major__) && __clang_major__ >= 23
  si8 z8 = {0, 0, 0, 0, 0, 0, 0, 0};
  __builtin_amdgcn_tensor_load_to_lds(g0, g1, z4, z4, z8, 0);
#else
  __builtin_amdgcn_tensor_load_to_lds(g0, g1, z4, z4, 0);
#endif
}

// data_size=2B | pad_enable, pad every 64B by 4 dwords -> LDS row stride 40 bf16
#define TDM_CFG_PAD64   (0x00010000u | 0x00100000u | (3u << 22) | (3u << 25))
// data_size=2B | pad_enable, pad every 128B by 4 dwords -> LDS row stride 72 bf16
#define TDM_CFG_PAD128  (0x00010000u | 0x00100000u | (4u << 22) | (3u << 25))

// ---------------------------------------------------------------------------
// f32 -> bf16 bulk convert (vectorized, grid-stride)
// ---------------------------------------------------------------------------
__global__ void f32_to_bf16_vec(const float* __restrict__ src,
                                __bf16* __restrict__ dst, int n4) {
  int i = blockIdx.x * blockDim.x + threadIdx.x;
  int stride = gridDim.x * blockDim.x;
  for (; i < n4; i += stride) {
    float4 f = reinterpret_cast<const float4*>(src)[i];
    v4bf o = { (__bf16)f.x, (__bf16)f.y, (__bf16)f.z, (__bf16)f.w };
    reinterpret_cast<v4bf*>(dst)[i] = o;
  }
}

// ---------------------------------------------------------------------------
// qk GEMM (16384x1024 @ 1024x2048) + bias + elu+1 + RoPE epilogue.
// Block tile 128(M) x 128(N), K-step 32, TDM double-buffered LDS staging:
//   wave0 streams A tiles, wave1 streams B tiles, all 8 waves compute
//   2x4 16x16 WMMA accumulators each (256 v_wmma per wave).
// grid (16, 128), block 256.
// ---------------------------------------------------------------------------
__global__ void gemm_qk_rope(const __bf16* __restrict__ xb,
                             const __bf16* __restrict__ wb,
                             const float*  __restrict__ bias,
                             __bf16* __restrict__ qr,
                             __bf16* __restrict__ kr,
                             float*  __restrict__ ksum) {
  __shared__ __bf16 sA[2][128][40];   // 128 rows x 32 bf16 + TDM pad (16B-aligned rows)
  __shared__ __bf16 sB[2][128][40];
  const int lane = threadIdx.x & 31;
  const int wv   = threadIdx.x >> 5;
  const int li = lane & 15, hi = lane >> 4;
  const int wm = wv >> 1;             // 0..3 : 32-row M strip
  const int wn = wv & 1;              // 0..1 : 64-col N half
  const int mblock = blockIdx.y * 128;
  const int jblock = blockIdx.x * 128;

  v8f zero = {0.f,0.f,0.f,0.f,0.f,0.f,0.f,0.f};
  v8f acc[2][4];
  #pragma unroll
  for (int mt = 0; mt < 2; ++mt)
    #pragma unroll
    for (int et = 0; et < 4; ++et) acc[mt][et] = zero;

  // Prologue: stage k-step 0.
  if (wv == 0) {
    tdm_load_2d((uint32_t)(uintptr_t)&sA[0][0][0],
                xb + (size_t)mblock * C_, TDM_CFG_PAD64, 32, 128, C_);
  } else if (wv == 1) {
    tdm_load_2d((uint32_t)(uintptr_t)&sB[0][0][0],
                wb + (size_t)jblock * C_, TDM_CFG_PAD64, 32, 128, C_);
  }

  for (int ks = 0; ks < 32; ++ks) {
    const int buf = ks & 1;
    if (wv == 0) {
      if (ks + 1 < 32) {
        tdm_load_2d((uint32_t)(uintptr_t)&sA[buf ^ 1][0][0],
                    xb + (size_t)mblock * C_ + (ks + 1) * 32,
                    TDM_CFG_PAD64, 32, 128, C_);
        __builtin_amdgcn_s_wait_tensorcnt(1);
      } else {
        __builtin_amdgcn_s_wait_tensorcnt(0);
      }
    } else if (wv == 1) {
      if (ks + 1 < 32) {
        tdm_load_2d((uint32_t)(uintptr_t)&sB[buf ^ 1][0][0],
                    wb + (size_t)jblock * C_ + (ks + 1) * 32,
                    TDM_CFG_PAD64, 32, 128, C_);
        __builtin_amdgcn_s_wait_tensorcnt(1);
      } else {
        __builtin_amdgcn_s_wait_tensorcnt(0);
      }
    }
    __syncthreads();                       // tile `ks` visible to all waves

    v16bf afr[2];
    #pragma unroll
    for (int mt = 0; mt < 2; ++mt)
      afr[mt] = loadA16(&sA[buf][32*wm + 16*mt + li][8*hi]);
    v16bf bfr[4];
    #pragma unroll
    for (int et = 0; et < 4; ++et)
      bfr[et] = loadA16(&sB[buf][64*wn + 16*et + li][8*hi]);
    #pragma unroll
    for (int mt = 0; mt < 2; ++mt)
      #pragma unroll
      for (int et = 0; et < 4; ++et)
        acc[mt][et] = wmma_bf16(afr[mt], bfr[et], acc[mt][et]);

    __syncthreads();                       // done reading before DMA reuses buffer
  }

  // Epilogue: bias + elu+1 + RoPE (pair partner lives in adjacent lane).
  #pragma unroll
  for (int et = 0; et < 4; ++et) {
    const int j  = jblock + 64*wn + 16*et + li;
    const float bj = bias[j];
    const int ch = j & (C_ - 1);
    const int kp = ch >> 1;
    const float ang = __expf(-(float)kp * (9.210340371976184f / 512.0f));
    const float ct = __cosf(ang), st = __sinf(ang);
    const float sgn = (j & 1) ? st : -st;
    const bool isK = (j >= C_);
    #pragma unroll
    for (int mt = 0; mt < 2; ++mt) {
      #pragma unroll
      for (int r = 0; r < 8; ++r) {
        const int m = mblock + 32*wm + 16*mt + r + 8*hi;
        float t = acc[mt][et][r] + bj;
        float v = (t > 0.0f) ? (t + 1.0f) : __expf(t);  // elu(t)+1
        float p = __shfl_xor(v, 1, 32);                 // RoPE pair partner
        float rv = ct * v + sgn * p;
        size_t o = (size_t)m * C_ + ch;
        if (!isK) {
          qr[o] = (__bf16)rv;
        } else {
          kr[o] = (__bf16)rv;
          atomicAdd(&ksum[(m >> 12) * C_ + ch], rv);
        }
      }
    }
  }
}

// ---------------------------------------------------------------------------
// z[b,h,n] = 1 / (dot(q_rope_row, ksum)/N + 1e-6)   (rotation-invariance)
// ---------------------------------------------------------------------------
__global__ void z_kernel(const __bf16* __restrict__ qr,
                         const float*  __restrict__ ksum,
                         float* __restrict__ z) {
  const int t = blockIdx.x * blockDim.x + threadIdx.x;  // (b,h,n) flat
  const int n = t & (N_ - 1);
  const int h = (t >> 12) & (H_ - 1);
  const int b = t >> 16;
  const __bf16* qp = qr + ((size_t)(b * N_ + n)) * C_ + h * D_;
  const float* kp = ksum + b * C_ + h * D_;
  float acc = 0.f;
  #pragma unroll 8
  for (int d = 0; d < D_; ++d) acc += (float)qp[d] * kp[d];
  z[t] = 1.0f / (acc * (1.0f / N_) + 1e-6f);
}

// ---------------------------------------------------------------------------
// kv[b,h] = K_rope^T @ V / n  (64x4096 x 4096x64); TDM-staged 32-row tiles,
// double buffered; stored transposed (bf16) as kvT[b,h,e,d].
// grid 64 (= b*h), block 128 (4 waves, one 16-row dk strip per wave).
// ---------------------------------------------------------------------------
__global__ void kv_kernel(const __bf16* __restrict__ kr,
                          const __bf16* __restrict__ xb,
                          __bf16* __restrict__ kvT) {
  __shared__ __bf16 sK[2][32][72];   // 32 n-rows x 64 bf16 + TDM pad
  __shared__ __bf16 sV[2][32][72];
  const int bh = blockIdx.x;
  const int b = bh >> 4, h = bh & 15;
  const int tid = threadIdx.x;
  const int lane = tid & 31, wv = tid >> 5;
  const int li = lane & 15, hi = lane >> 4;
  const __bf16* gK = kr + ((size_t)b * N_) * C_ + h * D_;
  const __bf16* gV = xb + ((size_t)b * N_) * C_ + h * D_;

  v8f zero = {0.f,0.f,0.f,0.f,0.f,0.f,0.f,0.f};
  v8f acc[4];
  #pragma unroll
  for (int et = 0; et < 4; ++et) acc[et] = zero;

  if (wv == 0) {
    tdm_load_2d((uint32_t)(uintptr_t)&sK[0][0][0], gK, TDM_CFG_PAD128, 64, 32, C_);
    tdm_load_2d((uint32_t)(uintptr_t)&sV[0][0][0], gV, TDM_CFG_PAD128, 64, 32, C_);
  }

  const int dk = 16 * wv + li;
  for (int it = 0; it < N_ / 32; ++it) {
    const int buf = it & 1;
    if (wv == 0) {
      if (it + 1 < N_ / 32) {
        const size_t goff = (size_t)(it + 1) * 32 * C_;
        tdm_load_2d((uint32_t)(uintptr_t)&sK[buf ^ 1][0][0], gK + goff,
                    TDM_CFG_PAD128, 64, 32, C_);
        tdm_load_2d((uint32_t)(uintptr_t)&sV[buf ^ 1][0][0], gV + goff,
                    TDM_CFG_PAD128, 64, 32, C_);
        __builtin_amdgcn_s_wait_tensorcnt(2);
      } else {
        __builtin_amdgcn_s_wait_tensorcnt(0);
      }
    }
    __syncthreads();

    v16bf a;                              // A = K_rope^T (row=dk, col=n) via LDS gather
    #pragma unroll
    for (int e = 0; e < 8; ++e) a[e]     = sK[buf][8*hi + e][dk];
    #pragma unroll
    for (int e = 0; e < 8; ++e) a[8 + e] = sK[buf][16 + 8*hi + e][dk];

    #pragma unroll
    for (int et = 0; et < 4; ++et) {
      v16bf bf;                           // B = V (row=n, col=e)
      #pragma unroll
      for (int e = 0; e < 16; ++e) bf[e] = sV[buf][8*hi + e + 8*(e >> 3)][16*et + li];
      acc[et] = wmma_bf16(a, bf, acc[et]);
    }
    __syncthreads();
  }

  const float inv_n = 1.0f / N_;
  #pragma unroll
  for (int et = 0; et < 4; ++et) {
    const int eg = 16*et + li;
    __bf16* dst = kvT + ((size_t)bh * D_ + eg) * D_ + 16*wv + 8*hi;
    v8bf o;
    #pragma unroll
    for (int r = 0; r < 8; ++r) o[r] = (__bf16)(acc[et][r] * inv_n);
    *reinterpret_cast<v8bf*>(dst) = o;
  }
}

// ---------------------------------------------------------------------------
// out = (Q_rope @ kv) * z + depthwise_conv3(x) + lepe_b
// grid (256 n-tiles, 4 b), block 256 (8 waves); wave handles heads wv, wv+8.
// ---------------------------------------------------------------------------
__global__ void attn_out_kernel(const __bf16* __restrict__ qr,
                                const __bf16* __restrict__ kvT,
                                const float* __restrict__ z,
                                const float* __restrict__ x,
                                const float* __restrict__ lw,
                                const float* __restrict__ lb,
                                float* __restrict__ out) {
  const int b  = blockIdx.y;
  const int n0 = blockIdx.x * 16;
  const int lane = threadIdx.x & 31, wv = threadIdx.x >> 5;
  const int li = lane & 15, hi = lane >> 4;
  v8f zero = {0.f,0.f,0.f,0.f,0.f,0.f,0.f,0.f};

  for (int hh = 0; hh < 2; ++hh) {
    const int h = wv + 8 * hh;
    v8f acc[4];
    #pragma unroll
    for (int et = 0; et < 4; ++et) acc[et] = zero;

    const __bf16* aBase = qr + ((size_t)(b*N_ + n0 + li)) * C_ + h*D_ + 8*hi;
    #pragma unroll
    for (int ks = 0; ks < 2; ++ks) {
      const int k0 = 32 * ks;
      v16bf a = loadA16(aBase + k0);
      #pragma unroll
      for (int et = 0; et < 4; ++et) {
        const __bf16* bp = kvT + ((size_t)(b*H_ + h) * D_ + 16*et + li) * D_
                               + k0 + 8*hi;
        v16bf bf = loadA16(bp);   // kv column = kvT row
        acc[et] = wmma_bf16(a, bf, acc[et]);
      }
    }

    #pragma unroll
    for (int et = 0; et < 4; ++et) {
      const int ch = h * D_ + 16*et + li;
      const float w0 = lw[ch*3 + 0], w1 = lw[ch*3 + 1], w2 = lw[ch*3 + 2];
      const float bb = lb[ch];
      #pragma unroll
      for (int r = 0; r < 8; ++r) {
        const int n = n0 + r + 8*hi;
        const float zz = z[((size_t)(b*H_ + h)) * N_ + n];
        const size_t xo = ((size_t)(b*N_ + n)) * C_ + ch;
        float pe = bb + w1 * x[xo];
        if (n > 0)      pe += w0 * x[xo - C_];
        if (n < N_ - 1) pe += w2 * x[xo + C_];
        out[xo] = acc[et][r] * zz + pe;
      }
    }
  }
}

// ---------------------------------------------------------------------------
extern "C" void kernel_launch(void* const* d_in, const int* in_sizes, int n_in,
                              void* d_out, int out_size, void* d_ws, size_t ws_size,
                              hipStream_t stream) {
  const float* x   = (const float*)d_in[0];
  const float* Wqk = (const float*)d_in[1];
  const float* bqk = (const float*)d_in[2];
  const float* lw  = (const float*)d_in[3];
  const float* lb  = (const float*)d_in[4];
  float* out = (float*)d_out;

  // Workspace layout (bytes), ~101.6 MiB total.
  char* ws = (char*)d_ws;
  __bf16* xb   = (__bf16*)(ws);                   // x bf16        33554432
  __bf16* wb   = (__bf16*)(ws + 33554432);        // W bf16         4194304
  __bf16* qr   = (__bf16*)(ws + 37748736);        // q_rope bf16   33554432
  __bf16* kr   = (__bf16*)(ws + 71303168);        // k_rope bf16   33554432
  float*  ksum = (float*) (ws + 104857600);       // sum_n k_rope     16384
  float*  zbuf = (float*) (ws + 104873984);       // z              1048576
  __bf16* kvT  = (__bf16*)(ws + 105922560);       // kv^T bf16       524288

  hipMemsetAsync(ksum, 0, (size_t)B_ * C_ * sizeof(float), stream);
  f32_to_bf16_vec<<<2048, 256, 0, stream>>>(x,   xb, (B_*N_*C_) / 4);
  f32_to_bf16_vec<<<512,  256, 0, stream>>>(Wqk, wb, (NQK*C_) / 4);
  gemm_qk_rope<<<dim3(16, 128), 256, 0, stream>>>(xb, wb, bqk, qr, kr, ksum);
  z_kernel<<<1024, 256, 0, stream>>>(qr, ksum, zbuf);
  kv_kernel<<<64, 128, 0, stream>>>(kr, xb, kvT);
  attn_out_kernel<<<dim3(256, 4), 256, 0, stream>>>(qr, kvT, zbuf, x, lw, lb, out);
}
